// GQAttention_71133248356455
// MI455X (gfx1250) — matrix operand order repaired and checked by reading, not compiled, not run on hardware
//
#include <hip/hip_runtime.h>
#include <stdint.h>

#define T_SEQ   2048
#define DMODEL  2048
#define NH      32
#define NKV     8
#define HD      64

typedef __attribute__((ext_vector_type(16))) __bf16 v16bf;
typedef __attribute__((ext_vector_type(8)))  float  v8f;

union FragBF { uint4 q[2]; v16bf v; };

__device__ __forceinline__ unsigned short f2bf(float f) {
  union { float f; unsigned u; } v; v.f = f;
  unsigned r = v.u + 0x7fffu + ((v.u >> 16) & 1u);   // round-to-nearest-even
  return (unsigned short)(r >> 16);
}

__device__ __forceinline__ v8f wmma_bf(const FragBF& a, const FragBF& b, v8f c) {
  return __builtin_amdgcn_wmma_f32_16x16x32_bf16(false, a.v, false, b.v,
                                                 (short)0, c, false, false);
}

// --- CDNA5 async global->LDS copy (ASYNCcnt-tracked), per CDNA5 ISA ---
// LDS operand is a DS-style offset = low 32 bits of the generic pointer.
__device__ __forceinline__ uint32_t lds_off(const void* p) {
  return (uint32_t)(uintptr_t)p;
}
__device__ __forceinline__ void async_ld_b128(uint32_t ldsoff, const void* g) {
  asm volatile("global_load_async_to_lds_b128 %0, %1, off"
               :: "v"(ldsoff), "v"((unsigned long long)(uintptr_t)g)
               : "memory");
}
__device__ __forceinline__ void async_wait0() {
  asm volatile("s_wait_asynccnt 0x0" ::: "memory");
}

// ---------------- elementwise cast f32 -> bf16 ----------------
__global__ void cast_bf16_kernel(const float* __restrict__ in,
                                 unsigned short* __restrict__ out, int n) {
  int i = blockIdx.x * blockDim.x + threadIdx.x;
  if (i < n) out[i] = f2bf(in[i]);
}

// ---------------- cast + transpose: W[K][N] f32 -> Wt[N][K] bf16 ----------------
__global__ void transpose_cast_kernel(const float* __restrict__ W,
                                      unsigned short* __restrict__ Wt,
                                      int K, int N) {
  __shared__ float tile[32][33];
  const int n0 = blockIdx.x * 32, k0 = blockIdx.y * 32;
  const int tx = threadIdx.x, ty = threadIdx.y;
  for (int j = 0; j < 32; j += 8)
    tile[ty + j][tx] = W[(size_t)(k0 + ty + j) * N + n0 + tx];
  __syncthreads();
  for (int j = 0; j < 32; j += 8)
    Wt[(size_t)(n0 + ty + j) * K + k0 + tx] = f2bf(tile[tx][ty + j]);
}

// ---------------- bf16 WMMA GEMM: C[M][N] f32 = A[M][K] * Bt[N][K]^T ----------------
// 256 threads = 8 waves; block tile 128x128; wave tile 64x32; K-step 32.
// Double-buffered LDS fed by async global->LDS copies (overlap with WMMA).
__launch_bounds__(256)
__global__ void gemm_bf16_kernel(const unsigned short* __restrict__ A,
                                 const unsigned short* __restrict__ Bt,
                                 float* __restrict__ C,
                                 int M, int N, int K) {
  __shared__ __align__(16) unsigned short sA[2][128 * 40]; // 32 used + 8 pad halves
  __shared__ __align__(16) unsigned short sB[2][128 * 40];

  const int tid  = threadIdx.x;
  const int lane = tid & 31, wid = tid >> 5;
  const int wm = wid >> 2, wn = wid & 3;           // 2x4 wave grid
  const int m0 = blockIdx.y * 128, n0 = blockIdx.x * 128;
  const int lm = lane & 15, half = lane >> 4;

  v8f acc[4][2];
  v8f vz = {0.f,0.f,0.f,0.f,0.f,0.f,0.f,0.f};
  for (int mi = 0; mi < 4; mi++)
    for (int ni = 0; ni < 2; ni++) acc[mi][ni] = vz;

  const int r = tid >> 1, sg = tid & 1;            // 2 threads per 32-half row
  const size_t aRow = (size_t)(m0 + r) * K;
  const size_t bRow = (size_t)(n0 + r) * K;

  auto issue_tile = [&](int it) {
    const int buf = it & 1;
    const int kk  = it * 32;
    const unsigned short* ga = A  + aRow + kk + sg * 16;
    const unsigned short* gb = Bt + bRow + kk + sg * 16;
    unsigned short* la = &sA[buf][r * 40 + sg * 16];
    unsigned short* lb = &sB[buf][r * 40 + sg * 16];
    async_ld_b128(lds_off(la),     ga);
    async_ld_b128(lds_off(la + 8), ga + 8);
    async_ld_b128(lds_off(lb),     gb);
    async_ld_b128(lds_off(lb + 8), gb + 8);
  };

  const int ntiles = K >> 5;
  issue_tile(0);

  for (int it = 0; it < ntiles; ++it) {
    async_wait0();                                  // tile `it` landed in LDS
    __syncthreads();                                // everyone done with tile it-1
    if (it + 1 < ntiles) issue_tile(it + 1);        // stream next tile during WMMA

    const unsigned short* bufA = sA[it & 1];
    const unsigned short* bufB = sB[it & 1];

    FragBF afr[4], bfr[2];
    for (int mi = 0; mi < 4; mi++) {                // A: lane=row, k = 8*half.. / 16+8*half..
      const unsigned short* p = bufA + (wm * 64 + mi * 16 + lm) * 40 + half * 8;
      afr[mi].q[0] = *(const uint4*)p;
      afr[mi].q[1] = *(const uint4*)(p + 16);
    }
    for (int ni = 0; ni < 2; ni++) {                // B: lane=col, k = 8*half.. / 16+8*half..
      const unsigned short* p = bufB + (wn * 32 + ni * 16 + lm) * 40 + half * 8;
      bfr[ni].q[0] = *(const uint4*)p;
      bfr[ni].q[1] = *(const uint4*)(p + 16);
    }
    for (int mi = 0; mi < 4; mi++)
      for (int ni = 0; ni < 2; ni++)
        acc[mi][ni] = wmma_bf(afr[mi], bfr[ni], acc[mi][ni]);
  }

  for (int mi = 0; mi < 4; mi++)
    for (int ni = 0; ni < 2; ni++)
      for (int rr = 0; rr < 8; rr++)
        C[(size_t)(m0 + wm * 64 + mi * 16 + half * 8 + rr) * N +
          (n0 + wn * 32 + ni * 16 + lm)] = acc[mi][ni][rr];
}

// ---------------- RoPE (interleaved pairs) + scale + layout to head-major bf16 ----------------
__global__ void rope_kernel(const float* __restrict__ X,     // [T][nheads][64] f32
                            const float* __restrict__ freqs, // [2][T][64]
                            unsigned short* __restrict__ out, // [nheads][T][64] bf16
                            int nheads, float scale) {
  int idx = blockIdx.x * blockDim.x + threadIdx.x;
  if (idx >= T_SEQ * nheads * (HD / 2)) return;
  int i  = idx & 31;
  int hh = (idx >> 5) % nheads;
  int t  = idx / (32 * nheads);
  size_t src = ((size_t)t * nheads + hh) * HD + 2 * i;
  float q0 = X[src], q1 = X[src + 1];
  float c = freqs[(size_t)t * HD + 2 * i];
  float s = freqs[(size_t)T_SEQ * HD + (size_t)t * HD + 2 * i];
  float o0 = (q0 * c - q1 * s) * scale;
  float o1 = (q1 * c + q0 * s) * scale;
  size_t dst = ((size_t)hh * T_SEQ + t) * HD + 2 * i;
  out[dst] = f2bf(o0);
  out[dst + 1] = f2bf(o1);
}

// ---------------- V: [T][KV][64] f32 -> Vt [KV][64][T] bf16 ----------------
__global__ void vtrans_kernel(const float* __restrict__ Vf,
                              unsigned short* __restrict__ Vt) {
  int idx = blockIdx.x * blockDim.x + threadIdx.x;   // NKV*HD*T
  int t  = idx & (T_SEQ - 1);
  int d  = (idx >> 11) & 63;
  int kv = idx >> 17;
  Vt[((size_t)kv * HD + d) * T_SEQ + t] = f2bf(Vf[((size_t)t * NKV + kv) * HD + d]);
}

// ---------------- flash attention: one 64-query tile per head per block ----------------
// 128 threads = 4 waves; each wave owns a 16-query strip, full d=64 output.
// K/V tiles double-buffered and streamed with async global->LDS copies.
__launch_bounds__(128)
__global__ void attn_kernel(const unsigned short* __restrict__ Qb,  // [H][T][64]
                            const unsigned short* __restrict__ Kb,  // [KV][T][64]
                            const unsigned short* __restrict__ Vtb, // [KV][64][T]
                            unsigned short* __restrict__ Ob) {      // [T][H*64]
  __shared__ __align__(16) unsigned short sQ[64 * 72];
  __shared__ __align__(16) unsigned short sK[2][64 * 72];
  __shared__ __align__(16) unsigned short sV[2][64 * 72];   // [dim][key]
  __shared__ __align__(16) float          sS[4][16 * 64];
  __shared__ __align__(16) unsigned short sP[4][16 * 72];
  __shared__ float sStat[64];

  const int qt = blockIdx.x, h = blockIdx.y;
  const int kvh = h >> 2;                                // G = 4
  const int tid = threadIdx.x, lane = tid & 31, wid = tid >> 5;
  const int lm = lane & 15, half = lane >> 4;

  auto issue_kv = [&](int kt) {
    const int buf = kt & 1;
    for (int idx = tid; idx < 512; idx += 128) {
      int rr = idx >> 3, sg = idx & 7;
      async_ld_b128(lds_off(&sK[buf][rr * 72 + sg * 8]),
                    Kb + ((size_t)kvh * T_SEQ + kt * 64 + rr) * HD + sg * 8);
      async_ld_b128(lds_off(&sV[buf][rr * 72 + sg * 8]),
                    Vtb + ((size_t)kvh * HD + rr) * T_SEQ + kt * 64 + sg * 8);
    }
  };

  // Q tile (async) + first K/V tile
  for (int idx = tid; idx < 512; idx += 128) {
    int rr = idx >> 3, sg = idx & 7;
    async_ld_b128(lds_off(&sQ[rr * 72 + sg * 8]),
                  Qb + ((size_t)h * T_SEQ + qt * 64 + rr) * HD + sg * 8);
  }
  issue_kv(0);
  async_wait0();
  __syncthreads();

  FragBF aq[2];
  for (int c = 0; c < 2; c++) {
    const unsigned short* p = sQ + (wid * 16 + lm) * 72 + c * 32 + half * 8;
    aq[c].q[0] = *(const uint4*)p;
    aq[c].q[1] = *(const uint4*)(p + 16);
  }

  v8f vz = {0.f,0.f,0.f,0.f,0.f,0.f,0.f,0.f};
  v8f o[4]; for (int ni = 0; ni < 4; ni++) o[ni] = vz;
  float row_max = -3.0e38f, row_sum = 0.0f;

  for (int kt = 0; kt <= qt; kt++) {
    if (kt > 0) {
      async_wait0();                                 // tile kt landed
      __syncthreads();                               // all waves done with kt-1
    }
    if (kt + 1 <= qt) issue_kv(kt + 1);              // stream next tile during compute

    const unsigned short* bK = sK[kt & 1];
    const unsigned short* bV = sV[kt & 1];

    // S strip = Q(16x64) . K^T(64x64): 4 tiles x 2 k-chunks
    for (int ni = 0; ni < 4; ni++) {
      FragBF bk0, bk1;
      const unsigned short* p0 = bK + (ni * 16 + lm) * 72 + half * 8;
      bk0.q[0] = *(const uint4*)p0;        bk0.q[1] = *(const uint4*)(p0 + 16);
      bk1.q[0] = *(const uint4*)(p0 + 32); bk1.q[1] = *(const uint4*)(p0 + 48);
      v8f s = wmma_bf(aq[0], bk0, vz);
      s     = wmma_bf(aq[1], bk1, s);
      for (int rr = 0; rr < 8; rr++)
        sS[wid][(half * 8 + rr) * 64 + ni * 16 + lm] = s[rr];
    }
    __syncthreads();

    // online softmax: one lane per strip row
    if (lane < 16) {
      int q_local = wid * 16 + lane;
      int vc = (kt == qt) ? (q_local + 1) : 64;      // causal limit
      const float* srow = &sS[wid][lane * 64];
      float mx = -3.0e38f;
      for (int j = 0; j < 64; j += 4) {
        float4 v = *(const float4*)(srow + j);
        if (j + 0 < vc) mx = fmaxf(mx, v.x);
        if (j + 1 < vc) mx = fmaxf(mx, v.y);
        if (j + 2 < vc) mx = fmaxf(mx, v.z);
        if (j + 3 < vc) mx = fmaxf(mx, v.w);
      }
      float nm = fmaxf(row_max, mx);
      float corr = __expf(row_max - nm);
      float ps = 0.0f;
      unsigned* prow = (unsigned*)&sP[wid][lane * 72];
      for (int j = 0; j < 64; j += 4) {
        float4 v = *(const float4*)(srow + j);
        float p0 = (j + 0 < vc) ? __expf(v.x - nm) : 0.0f;
        float p1 = (j + 1 < vc) ? __expf(v.y - nm) : 0.0f;
        float p2 = (j + 2 < vc) ? __expf(v.z - nm) : 0.0f;
        float p3 = (j + 3 < vc) ? __expf(v.w - nm) : 0.0f;
        prow[(j >> 1)]     = (unsigned)f2bf(p0) | ((unsigned)f2bf(p1) << 16);
        prow[(j >> 1) + 1] = (unsigned)f2bf(p2) | ((unsigned)f2bf(p3) << 16);
        ps += (p0 + p1) + (p2 + p3);
      }
      row_sum = row_sum * corr + ps;
      row_max = nm;
      sStat[wid * 16 + lane] = corr;
    }
    __syncthreads();

    float cr[8];
    for (int rr = 0; rr < 8; rr++) cr[rr] = sStat[wid * 16 + half * 8 + rr];
    for (int ni = 0; ni < 4; ni++)
      for (int rr = 0; rr < 8; rr++) o[ni][rr] *= cr[rr];

    FragBF ap[2];
    for (int c = 0; c < 2; c++) {
      const unsigned short* p = sP[wid] + lm * 72 + c * 32 + half * 8;
      ap[c].q[0] = *(const uint4*)p;
      ap[c].q[1] = *(const uint4*)(p + 16);
    }
    for (int ni = 0; ni < 4; ni++) {                     // O += P(16x64) . V(64x64)
      FragBF bv0, bv1;
      const unsigned short* p0 = bV + (ni * 16 + lm) * 72 + half * 8;
      bv0.q[0] = *(const uint4*)p0;        bv0.q[1] = *(const uint4*)(p0 + 16);
      bv1.q[0] = *(const uint4*)(p0 + 32); bv1.q[1] = *(const uint4*)(p0 + 48);
      o[ni] = wmma_bf(ap[0], bv0, o[ni]);
      o[ni] = wmma_bf(ap[1], bv1, o[ni]);
    }
  }

  if (lane < 16) sStat[wid * 16 + lane] = 1.0f / row_sum;
  __syncthreads();
  float inv[8];
  for (int rr = 0; rr < 8; rr++) inv[rr] = sStat[wid * 16 + half * 8 + rr];
  for (int ni = 0; ni < 4; ni++)
    for (int rr = 0; rr < 8; rr++) {
      size_t row = (size_t)qt * 64 + wid * 16 + half * 8 + rr;
      Ob[row * DMODEL + h * 64 + ni * 16 + lm] = f2bf(o[ni][rr] * inv[rr]);
    }
}

// ============================================================================
extern "C" void kernel_launch(void* const* d_in, const int* in_sizes, int n_in,
                              void* d_out, int out_size, void* d_ws, size_t ws_size,
                              hipStream_t stream) {
  (void)in_sizes; (void)n_in; (void)out_size; (void)ws_size;
  const float* x     = (const float*)d_in[0];
  const float* freqs = (const float*)d_in[1];
  // d_in[2] (mask) unused: causality computed analytically
  const float* Wq = (const float*)d_in[3];
  const float* Wk = (const float*)d_in[4];
  const float* Wv = (const float*)d_in[5];
  const float* Wo = (const float*)d_in[6];
  float* out = (float*)d_out;

  char* ws = (char*)d_ws;
  size_t off = 0;
  auto alloc = [&](size_t bytes) { size_t o = off; off = (off + bytes + 255) & ~(size_t)255; return o; };

  unsigned short* xbf  = (unsigned short*)(ws + alloc((size_t)T_SEQ * DMODEL * 2));
  unsigned short* wqT  = (unsigned short*)(ws + alloc((size_t)DMODEL * DMODEL * 2));
  unsigned short* wkT  = (unsigned short*)(ws + alloc((size_t)(NKV * HD) * DMODEL * 2));
  unsigned short* wvT  = (unsigned short*)(ws + alloc((size_t)(NKV * HD) * DMODEL * 2));
  unsigned short* woT  = (unsigned short*)(ws + alloc((size_t)DMODEL * DMODEL * 2));
  float*          Qf   = (float*)(ws + alloc((size_t)T_SEQ * NH * HD * 4));
  float*          Kf   = (float*)(ws + alloc((size_t)T_SEQ * NKV * HD * 4));
  float*          Vf   = (float*)(ws + alloc((size_t)T_SEQ * NKV * HD * 4));
  unsigned short* Qbf  = (unsigned short*)(ws + alloc((size_t)NH * T_SEQ * HD * 2));
  unsigned short* Kbf  = (unsigned short*)(ws + alloc((size_t)NKV * T_SEQ * HD * 2));
  unsigned short* Vtbf = (unsigned short*)(ws + alloc((size_t)NKV * HD * T_SEQ * 2));
  unsigned short* Aobf = (unsigned short*)(ws + alloc((size_t)T_SEQ * DMODEL * 2));

  // 1) casts / transposed-cast of weights
  {
    int n = T_SEQ * DMODEL;
    cast_bf16_kernel<<<n / 256, 256, 0, stream>>>(x, xbf, n);
  }
  transpose_cast_kernel<<<dim3(DMODEL / 32, DMODEL / 32), dim3(32, 8), 0, stream>>>(Wq, wqT, DMODEL, DMODEL);
  transpose_cast_kernel<<<dim3((NKV * HD) / 32, DMODEL / 32), dim3(32, 8), 0, stream>>>(Wk, wkT, DMODEL, NKV * HD);
  transpose_cast_kernel<<<dim3((NKV * HD) / 32, DMODEL / 32), dim3(32, 8), 0, stream>>>(Wv, wvT, DMODEL, NKV * HD);
  transpose_cast_kernel<<<dim3(DMODEL / 32, DMODEL / 32), dim3(32, 8), 0, stream>>>(Wo, woT, DMODEL, DMODEL);

  // 2) projections (bf16 WMMA GEMM, fp32 out)
  gemm_bf16_kernel<<<dim3(DMODEL / 128, T_SEQ / 128), 256, 0, stream>>>(xbf, wqT, Qf, T_SEQ, DMODEL, DMODEL);
  gemm_bf16_kernel<<<dim3((NKV * HD) / 128, T_SEQ / 128), 256, 0, stream>>>(xbf, wkT, Kf, T_SEQ, NKV * HD, DMODEL);
  gemm_bf16_kernel<<<dim3((NKV * HD) / 128, T_SEQ / 128), 256, 0, stream>>>(xbf, wvT, Vf, T_SEQ, NKV * HD, DMODEL);

  // 3) RoPE (scale 1/sqrt(64) folded into Q) and V transpose
  {
    int nq = T_SEQ * NH * (HD / 2);
    rope_kernel<<<nq / 256, 256, 0, stream>>>(Qf, freqs, Qbf, NH, 0.125f);
    int nk = T_SEQ * NKV * (HD / 2);
    rope_kernel<<<nk / 256, 256, 0, stream>>>(Kf, freqs, Kbf, NKV, 1.0f);
    int nv = NKV * HD * T_SEQ;
    vtrans_kernel<<<nv / 256, 256, 0, stream>>>(Vf, Vtbf);
  }

  // 4) flash attention
  attn_kernel<<<dim3(T_SEQ / 64, NH), 128, 0, stream>>>(Qbf, Kbf, Vtbf, Aobf);

  // 5) output projection -> d_out (fp32)
  gemm_bf16_kernel<<<dim3(DMODEL / 128, T_SEQ / 128), 256, 0, stream>>>(Aobf, woT, out, T_SEQ, DMODEL, DMODEL);
}
